// PSI_47931835024047
// MI455X (gfx1250) — compile-verified, matrix operand order checked by reading, not run on hardware
//
#include <hip/hip_runtime.h>
#include <hip/hip_bf16.h>
#include <math.h>

// ---------------------------------------------------------------------------
// Types for CDNA5 WMMA (wave32): v_wmma_f32_16x16x32_bf16
// ---------------------------------------------------------------------------
typedef __attribute__((ext_vector_type(16))) __bf16 v16bf;
typedef __attribute__((ext_vector_type(8)))  __bf16 v8bf;
typedef __attribute__((ext_vector_type(8)))  float  v8f;
typedef int vsi4 __attribute__((vector_size(16)));   // int4 for async builtin

#define AS1 __attribute__((address_space(1)))
#define AS3 __attribute__((address_space(3)))

#if defined(__has_builtin)
#  if __has_builtin(__builtin_amdgcn_global_load_async_to_lds_b128)
#    define HAVE_ASYNC 1
#  endif
#  if __has_builtin(__builtin_amdgcn_s_wait_asynccnt)
#    define HAVE_WAITA 1
#  endif
#endif
#ifndef HAVE_ASYNC
#  define HAVE_ASYNC 0
#endif
#ifndef HAVE_WAITA
#  define HAVE_WAITA 0
#endif

// Async global->LDS 16-byte copy (ASYNCcnt path) with synchronous fallback.
static __device__ __forceinline__ void async_cp16(const __bf16* g, __bf16* l) {
#if HAVE_ASYNC
    __builtin_amdgcn_global_load_async_to_lds_b128(
        (AS1 vsi4*)(g), (AS3 vsi4*)(l), 0, 0);
#else
    *(v8bf*)l = *(const v8bf*)g;
#endif
}
static __device__ __forceinline__ void wait_async0() {
#if HAVE_ASYNC
#  if HAVE_WAITA
    __builtin_amdgcn_s_wait_asynccnt(0);
#  else
    asm volatile("s_wait_asynccnt 0" ::: "memory");
#  endif
#endif
}

static __device__ __forceinline__ __bf16 f2bf(float f) {
    union { float f; unsigned u; } cv; cv.f = f;
    unsigned u = cv.u;
    u += 0x7FFFu + ((u >> 16) & 1u);          // round-to-nearest-even
    unsigned short h = (unsigned short)(u >> 16);
    union { unsigned short s; __bf16 b; } o; o.s = h;
    return o.b;
}
static __device__ __forceinline__ float bf2f(__bf16 b) {
    union { __bf16 b; unsigned short s; } i; i.b = b;
    union { unsigned u; float f; } o; o.u = ((unsigned)i.s) << 16;
    return o.f;
}

static __device__ __forceinline__ v8f zero8() {
    v8f z;
#pragma unroll
    for (int i = 0; i < 8; ++i) z[i] = 0.0f;
    return z;
}

// Fragment loader.  16-bit A (16x32) / B^T (16x32) layout, wave32:
//   lane L: row = L&15, hi = L>>4
//   element e<8 : k = hi*8 + e          -> halves [hi*8,   hi*8+8)
//   element e>=8: k = hi*8 + e + 8      -> halves [hi*8+16, hi*8+24)
static __device__ __forceinline__ v16bf ldfrag(const __bf16* rowp, int hi) {
    v8bf lo  = *(const v8bf*)(rowp + hi * 8);
    v8bf hi2 = *(const v8bf*)(rowp + hi * 8 + 16);
    v16bf r;
#pragma unroll
    for (int i = 0; i < 8; ++i) { r[i] = lo[i]; r[i + 8] = hi2[i]; }
    return r;
}

static __device__ __forceinline__ v8f wmma_bf16(v16bf a, v16bf b, v8f c) {
    return __builtin_amdgcn_wmma_f32_16x16x32_bf16(false, a, false, b,
                                                   (short)0, c, false, false);
}

#define DIMD 512
#define NCTX 2048   // 4*D
#define NH   1024   // 2*D

// ---------------------------------------------------------------------------
// Weight prep: f32 [K][N] row-major  ->  bf16 [N][K] (transposed)
// ---------------------------------------------------------------------------
__global__ void __launch_bounds__(256)
PSI_prep_wt(const float* __restrict__ src, __bf16* __restrict__ dst,
            int K, int N) {
    int idx = blockIdx.x * 256 + threadIdx.x;
    if (idx < K * N) {
        int n = idx / K;
        int k = idx - n * K;
        dst[idx] = f2bf(src[k * N + n]);
    }
}

// ---------------------------------------------------------------------------
// Kernel 1: per-chunk fused  omega GEMM + phi cumsum + sincos recurrence
//           + LN statistics + in-place LayerNorm of ctx.
// One 256-thread block (8 waves) per 64-row chunk.
// Dynamic LDS:
//   As   : 2 x 64 x 40 bf16 (double-buffered x tile)  10240 B
//   phi  : 64 x 516 f32                               132096 B
//   rsum : 64 f32, rsumsq : 64 f32                    512 B
// ---------------------------------------------------------------------------
#define K1_ASTR 40
#define K1_PSTR 516
#define K1_SMEM (2 * 64 * K1_ASTR * 2 + 64 * K1_PSTR * 4 + 64 * 4 * 2)

__global__ void __launch_bounds__(256)
PSI_chunk(const float* __restrict__ x,
          const __bf16* __restrict__ Wot,       // [512][512] bf16 (N-major)
          const float* __restrict__ b_omega,
          const float* __restrict__ log_scale,
          const float* __restrict__ gamma,      // [2048]
          const float* __restrict__ beta,       // [2048]
          __bf16* __restrict__ ctx) {           // [16384][2048] bf16 (normed)
    extern __shared__ char smem[];
    __bf16* As   = (__bf16*)smem;                       // 2 buffers
    float*  phi  = (float*)(smem + 2 * 64 * K1_ASTR * 2);
    float*  rsum = phi + 64 * K1_PSTR;
    float*  rsq  = rsum + 64;

    const int t = threadIdx.x;
    const int lane = t & 31, w = t >> 5;
    const int l15 = lane & 15, hi = lane >> 4;
    const long row0 = (long)blockIdx.x * 64;

    // ---- GEMM: omega[64][512] = x_chunk @ W_omega ----
    v8f acc[4][4];
#pragma unroll
    for (int ns = 0; ns < 4; ++ns)
#pragma unroll
        for (int i = 0; i < 4; ++i) acc[ns][i] = zero8();

    const int sr  = t >> 2;            // staging row 0..63
    const int sc8 = (t & 3) * 8;       // staging col {0,8,16,24}

    // prologue: stage K-step 0 into buffer 0  (f32 -> bf16 conversion)
    {
        const float* src = x + (row0 + sr) * DIMD + sc8;
        float4 f0 = *(const float4*)(src);
        float4 f1 = *(const float4*)(src + 4);
        v8bf v;
        v[0] = f2bf(f0.x); v[1] = f2bf(f0.y); v[2] = f2bf(f0.z); v[3] = f2bf(f0.w);
        v[4] = f2bf(f1.x); v[5] = f2bf(f1.y); v[6] = f2bf(f1.z); v[7] = f2bf(f1.w);
        *(v8bf*)(As + sr * K1_ASTR + sc8) = v;
    }
    __syncthreads();

    for (int kt = 0; kt < DIMD / 32; ++kt) {
        const int k0  = kt * 32;
        const int cur = kt & 1;
        if (kt + 1 < DIMD / 32) {   // stage next K-step into other buffer
            const float* src = x + (row0 + sr) * DIMD + (k0 + 32) + sc8;
            float4 f0 = *(const float4*)(src);
            float4 f1 = *(const float4*)(src + 4);
            v8bf v;
            v[0] = f2bf(f0.x); v[1] = f2bf(f0.y); v[2] = f2bf(f0.z); v[3] = f2bf(f0.w);
            v[4] = f2bf(f1.x); v[5] = f2bf(f1.y); v[6] = f2bf(f1.z); v[7] = f2bf(f1.w);
            *(v8bf*)(As + (cur ^ 1) * 64 * K1_ASTR + sr * K1_ASTR + sc8) = v;
        }

        v16bf af[4];
#pragma unroll
        for (int i = 0; i < 4; ++i)
            af[i] = ldfrag(As + cur * 64 * K1_ASTR + (i * 16 + l15) * K1_ASTR, hi);

#pragma unroll
        for (int ns = 0; ns < 4; ++ns) {
            const int n = w * 64 + ns * 16 + l15;
            v16bf bf = ldfrag(Wot + (long)n * DIMD + k0, hi);
#pragma unroll
            for (int i = 0; i < 4; ++i)
                acc[ns][i] = wmma_bf16(af[i], bf, acc[ns][i]);
        }
        __syncthreads();
    }

    // ---- phi_raw = (omega + b) * exp(log_scale) / sqrt(pos) into LDS ----
    float bn[4], sn[4];
#pragma unroll
    for (int ns = 0; ns < 4; ++ns) {
        const int n = w * 64 + ns * 16 + l15;
        bn[ns] = b_omega[n];
        sn[ns] = expf(log_scale[n]);
    }
#pragma unroll
    for (int ns = 0; ns < 4; ++ns) {
        const int n = w * 64 + ns * 16 + l15;
#pragma unroll
        for (int i = 0; i < 4; ++i)
#pragma unroll
            for (int r = 0; r < 8; ++r) {
                const int m = i * 16 + r + 8 * hi;
                phi[m * K1_PSTR + n] =
                    (acc[ns][i][r] + bn[ns]) * sn[ns] * rsqrtf((float)(m + 1));
            }
    }
    if (t < 64) { rsum[t] = 0.0f; rsq[t] = 0.0f; }
    __syncthreads();

    // ---- cumsum over rows (thread owns cols t and t+256) ----
    {
        float run0 = 0.0f, run1 = 0.0f;
        for (int r = 0; r < 64; ++r) {
            run0 += phi[r * K1_PSTR + t];        phi[r * K1_PSTR + t]       = run0;
            run1 += phi[r * K1_PSTR + t + 256];  phi[r * K1_PSTR + t + 256] = run1;
        }
    }
    __syncthreads();

    // ---- sincos recurrence + raw ctx write (bf16) + LN partial sums ----
    float sr0 = 0, si0 = 0, sr1 = 0, si1 = 0;
    for (int r = 0; r < 64; ++r) {
        const float inv = 1.0f / (float)(r + 1);
        const float* xr = x + (row0 + r) * DIMD;
        __bf16* cb = ctx + (row0 + r) * (long)NCTX;
        float p1 = 0.0f, p2 = 0.0f;

        {   // col = t
            const int c = t;
            float xv = xr[c];
            float ph = phi[r * K1_PSTR + c];
            float s, co; __sincosf(ph, &s, &co);
            float cr = xv * co, ci = xv * s;
            sr0 += cr; si0 += ci;
            float mr = sr0 * inv, mi = si0 * inv;
            float rr = mr * co + mi * s;
            float ri = mi * co - mr * s;
            cb[c] = f2bf(cr); cb[512 + c] = f2bf(ci);
            cb[1024 + c] = f2bf(rr); cb[1536 + c] = f2bf(ri);
            p1 += cr + ci + rr + ri;
            p2 += cr * cr + ci * ci + rr * rr + ri * ri;
        }
        {   // col = t + 256
            const int c = t + 256;
            float xv = xr[c];
            float ph = phi[r * K1_PSTR + c];
            float s, co; __sincosf(ph, &s, &co);
            float cr = xv * co, ci = xv * s;
            sr1 += cr; si1 += ci;
            float mr = sr1 * inv, mi = si1 * inv;
            float rr = mr * co + mi * s;
            float ri = mi * co - mr * s;
            cb[c] = f2bf(cr); cb[512 + c] = f2bf(ci);
            cb[1024 + c] = f2bf(rr); cb[1536 + c] = f2bf(ri);
            p1 += cr + ci + rr + ri;
            p2 += cr * cr + ci * ci + rr * rr + ri * ri;
        }
#pragma unroll
        for (int o = 16; o > 0; o >>= 1) {
            p1 += __shfl_xor(p1, o, 32);
            p2 += __shfl_xor(p2, o, 32);
        }
        if (lane == 0) { atomicAdd(&rsum[r], p1); atomicAdd(&rsq[r], p2); }
    }
    // make this block's ctx stores visible to all its threads
    asm volatile("s_wait_storecnt 0" ::: "memory");
    __syncthreads();

    if (t < 64) {   // convert partials -> mu / rstd, kept in LDS
        float mu  = rsum[t] * (1.0f / (float)NCTX);
        float var = rsq[t] * (1.0f / (float)NCTX) - mu * mu;
        rsum[t] = mu;
        rsq[t]  = rsqrtf(var + 1e-5f);
    }
    __syncthreads();

    // ---- in-place LayerNorm of this chunk's ctx (L2-hot) ----
    {
        const int cbase = t * 8;                   // 2048 = 256 * 8
        float g[8], be[8];
#pragma unroll
        for (int i = 0; i < 8; ++i) { g[i] = gamma[cbase + i]; be[i] = beta[cbase + i]; }
        for (int r = 0; r < 64; ++r) {
            const float muv = rsum[r], rsv = rsq[r];
            __bf16* p = ctx + (row0 + r) * (long)NCTX + cbase;
            v8bf v = *(const v8bf*)p;
            v8bf o;
#pragma unroll
            for (int i = 0; i < 8; ++i)
                o[i] = f2bf((bf2f(v[i]) - muv) * rsv * g[i] + be[i]);
            *(v8bf*)p = o;
        }
    }
}

// ---------------------------------------------------------------------------
// Kernel 2: h = gelu( ctx_norm @ W1 + b1 )
// Block = 256 threads (8 waves), tile = 128(M) x 128(N), K step = 32,
// double-buffered LDS, async global->LDS staging.
// ---------------------------------------------------------------------------
__global__ void __launch_bounds__(256)
PSI_gemm_gelu(const __bf16* __restrict__ ctx,   // [16384][2048] normed bf16
              const __bf16* __restrict__ W1t,   // [1024][2048] bf16
              const float* __restrict__ b1,
              __bf16* __restrict__ hout) {
    const int K = NCTX, N = NH, KS = NCTX / 32;
    __shared__ __align__(16) __bf16 As[2][128 * 40];
    __shared__ __align__(16) __bf16 Bs[2][128 * 40];

    const int t = threadIdx.x;
    const int lane = t & 31, w = t >> 5;
    const int l15 = lane & 15, hi = lane >> 4;
    const long m0 = (long)blockIdx.x * 128;
    const int  n0 = blockIdx.y * 128;

    const int sr  = t >> 2;            // 0..63
    const int sc8 = (t & 3) * 8;

    v8f acc[8];
#pragma unroll
    for (int ns = 0; ns < 8; ++ns) acc[ns] = zero8();

    // prologue: stage K-step 0
#pragma unroll
    for (int i = 0; i < 2; ++i) {
        async_cp16(ctx + (m0 + sr + i * 64) * (long)K + sc8,
                   &As[0][(sr + i * 64) * 40 + sc8]);
        async_cp16(W1t + (long)(n0 + sr + i * 64) * K + sc8,
                   &Bs[0][(sr + i * 64) * 40 + sc8]);
    }
    wait_async0();
    __syncthreads();

    for (int kt = 0; kt < KS; ++kt) {
        const int k0 = kt * 32;
        const int cur = kt & 1;
        if (kt + 1 < KS) {
#pragma unroll
            for (int i = 0; i < 2; ++i) {
                async_cp16(ctx + (m0 + sr + i * 64) * (long)K + (k0 + 32) + sc8,
                           &As[cur ^ 1][(sr + i * 64) * 40 + sc8]);
                async_cp16(W1t + (long)(n0 + sr + i * 64) * K + (k0 + 32) + sc8,
                           &Bs[cur ^ 1][(sr + i * 64) * 40 + sc8]);
            }
            __builtin_prefetch(W1t + (long)(n0 + sr) * K + (k0 + 64) + sc8, 0, 1);
        }

        v16bf a = ldfrag(&As[cur][(w * 16 + l15) * 40], hi);
#pragma unroll
        for (int ns = 0; ns < 8; ++ns) {
            v16bf b = ldfrag(&Bs[cur][(ns * 16 + l15) * 40], hi);
            acc[ns] = wmma_bf16(a, b, acc[ns]);
        }
        wait_async0();
        __syncthreads();
    }

#pragma unroll
    for (int ns = 0; ns < 8; ++ns) {
        const int n = n0 + ns * 16 + l15;
        const float bias = b1[n];
#pragma unroll
        for (int r = 0; r < 8; ++r) {
            const long m = m0 + w * 16 + r + 8 * hi;
            float v = acc[ns][r] + bias;
            float ge = 0.5f * v * (1.0f + erff(v * 0.70710678118f));
            hout[m * (long)N + n] = f2bf(ge);
        }
    }
}

// ---------------------------------------------------------------------------
// Kernel 3: out = x + h @ W2 + b2    (f32 output, residual fused)
// Same skeleton: 128x128 tile, double-buffered async staging.
// ---------------------------------------------------------------------------
__global__ void __launch_bounds__(256)
PSI_gemm_res(const __bf16* __restrict__ h,      // [16384][1024] bf16
             const __bf16* __restrict__ W2t,    // [512][1024] bf16
             const float* __restrict__ b2,
             const float* __restrict__ x,
             float* __restrict__ out) {
    const int K = NH, N = DIMD, KS = NH / 32;
    __shared__ __align__(16) __bf16 As[2][128 * 40];
    __shared__ __align__(16) __bf16 Bs[2][128 * 40];

    const int t = threadIdx.x;
    const int lane = t & 31, w = t >> 5;
    const int l15 = lane & 15, hi = lane >> 4;
    const long m0 = (long)blockIdx.x * 128;
    const int  n0 = blockIdx.y * 128;

    const int sr  = t >> 2;
    const int sc8 = (t & 3) * 8;

    v8f acc[8];
#pragma unroll
    for (int ns = 0; ns < 8; ++ns) acc[ns] = zero8();

#pragma unroll
    for (int i = 0; i < 2; ++i) {
        async_cp16(h + (m0 + sr + i * 64) * (long)K + sc8,
                   &As[0][(sr + i * 64) * 40 + sc8]);
        async_cp16(W2t + (long)(n0 + sr + i * 64) * K + sc8,
                   &Bs[0][(sr + i * 64) * 40 + sc8]);
    }
    wait_async0();
    __syncthreads();

    for (int kt = 0; kt < KS; ++kt) {
        const int k0 = kt * 32;
        const int cur = kt & 1;
        if (kt + 1 < KS) {
#pragma unroll
            for (int i = 0; i < 2; ++i) {
                async_cp16(h + (m0 + sr + i * 64) * (long)K + (k0 + 32) + sc8,
                           &As[cur ^ 1][(sr + i * 64) * 40 + sc8]);
                async_cp16(W2t + (long)(n0 + sr + i * 64) * K + (k0 + 32) + sc8,
                           &Bs[cur ^ 1][(sr + i * 64) * 40 + sc8]);
            }
        }

        v16bf a = ldfrag(&As[cur][(w * 16 + l15) * 40], hi);
#pragma unroll
        for (int ns = 0; ns < 8; ++ns) {
            v16bf b = ldfrag(&Bs[cur][(ns * 16 + l15) * 40], hi);
            acc[ns] = wmma_bf16(a, b, acc[ns]);
        }
        wait_async0();
        __syncthreads();
    }

#pragma unroll
    for (int ns = 0; ns < 8; ++ns) {
        const int n = n0 + ns * 16 + l15;
        const float bias = b2[n];
#pragma unroll
        for (int r = 0; r < 8; ++r) {
            const long m = m0 + w * 16 + r + 8 * hi;
            out[m * (long)N + n] = acc[ns][r] + bias + x[m * (long)N + n];
        }
    }
}

// ---------------------------------------------------------------------------
// Launch.  Workspace layout (bytes):
//   Wot  bf16 512x512    @ 0          (524288)
//   W1t  bf16 1024x2048  @ 524288     (4194304)
//   W2t  bf16 512x1024   @ 4718592    (1048576)
//   ctx  bf16 16384x2048 @ 5767168    (67108864)
//   h    bf16 16384x1024 @ 72876032   (33554432)   total ~101.5 MB
// ---------------------------------------------------------------------------
extern "C" void kernel_launch(void* const* d_in, const int* in_sizes, int n_in,
                              void* d_out, int out_size, void* d_ws, size_t ws_size,
                              hipStream_t stream) {
    const float* x         = (const float*)d_in[0];
    const float* W_omega   = (const float*)d_in[1];
    const float* b_omega   = (const float*)d_in[2];
    const float* log_scale = (const float*)d_in[3];
    const float* ln_gamma  = (const float*)d_in[4];
    const float* ln_beta   = (const float*)d_in[5];
    const float* W1        = (const float*)d_in[6];
    const float* b1        = (const float*)d_in[7];
    const float* W2        = (const float*)d_in[8];
    const float* b2        = (const float*)d_in[9];
    float* out = (float*)d_out;

    char* ws = (char*)d_ws;
    __bf16* Wot  = (__bf16*)(ws + 0);
    __bf16* W1t  = (__bf16*)(ws + 524288);
    __bf16* W2t  = (__bf16*)(ws + 4718592);
    __bf16* ctx  = (__bf16*)(ws + 5767168);
    __bf16* h    = (__bf16*)(ws + 72876032);

    PSI_prep_wt<<<(512 * 512 + 255) / 256, 256, 0, stream>>>(W_omega, Wot, 512, 512);
    PSI_prep_wt<<<(2048 * 1024 + 255) / 256, 256, 0, stream>>>(W1, W1t, 2048, 1024);
    PSI_prep_wt<<<(1024 * 512 + 255) / 256, 256, 0, stream>>>(W2, W2t, 1024, 512);

    PSI_chunk<<<256, 256, K1_SMEM, stream>>>(x, Wot, b_omega, log_scale,
                                             ln_gamma, ln_beta, ctx);

    PSI_gemm_gelu<<<dim3(128, 8), 256, 0, stream>>>(ctx, W1t, b1, h);

    PSI_gemm_res<<<dim3(128, 4), 256, 0, stream>>>(h, W2t, b2, x, out);
}
